// VanillaSelfAttention_88656714925489
// MI455X (gfx1250) — matrix operand order repaired and checked by reading, not compiled, not run on hardware
//
#include <hip/hip_runtime.h>
#include <hip/hip_bf16.h>

typedef __attribute__((ext_vector_type(16))) _Float16 v16h;
typedef __attribute__((ext_vector_type(8)))  _Float16 v8h;
typedef __attribute__((ext_vector_type(8)))  float    v8f;

// ---------------------------------------------------------------------------
// Problem constants (reference: B=8, C=384, heads=12, d=32, Nz=256, Nx=1024)
#define BATCH 8
#define CCH   384
#define NHEAD 12
#define DHEAD 32
#define NZ    256
#define NX    1024
#define NTOK  1280
#define QKVO  1536
#define SCALE 0.17677669529663687f  // 32^-0.5

// Workspace layout (bytes)
#define OFF_WQH  ((size_t)0)
#define SZ_WQH   ((size_t)QKVO*CCH*2)
#define OFF_WPH  (OFF_WQH + SZ_WQH)
#define SZ_WPH   ((size_t)CCH*CCH*2)
#define OFF_Q    (OFF_WPH + SZ_WPH)
#define SZ_QKV   ((size_t)BATCH*NHEAD*NTOK*DHEAD*2)
#define OFF_K    (OFF_Q + SZ_QKV)
#define OFF_VT   (OFF_K + SZ_QKV)
#define OFF_O    (OFF_VT + SZ_QKV)
#define SZ_O32   ((size_t)BATCH*CCH*NTOK*4)
#define OFF_AOUT (OFF_O + SZ_O32)
#define OFF_U    (OFF_AOUT + SZ_O32)

// ---------------------------------------------------------------------------
__device__ __forceinline__ v8f wmma_f16(v16h a, v16h b, v8f c) {
  return __builtin_amdgcn_wmma_f32_16x16x32_f16(false, a, false, b,
                                                (short)0, c, false, false);
}

__device__ __forceinline__ void lds_fence() {
  asm volatile("s_wait_dscnt 0" ::: "memory");
}

// A-operand (16x32 f16): lane -> row = l&15; lanes 0-15 hold K 0-15,
// lanes 16-31 hold K 16-31 (2 f16 per VGPR).
__device__ __forceinline__ v16h load_a16x32(const _Float16* base, int ld, int lane) {
  int row = lane & 15;
  int kb  = (lane >> 4) * 16;
  union { v16h v; v8h h8[2]; } u;
  u.h8[0] = *(const v8h*)(base + (size_t)row * ld + kb);
  u.h8[1] = *(const v8h*)(base + (size_t)row * ld + kb + 8);
  return u.v;
}

// B-operand (32x16 f16): lane -> col = l&15, K = (l>>4)*16 .. +15 contiguous
// from a per-column pointer (column-major-ish source, K contiguous).
__device__ __forceinline__ v16h load_bcol(const _Float16* colptr) {
  union { v16h v; v8h h8[2]; } u;
  u.h8[0] = *(const v8h*)(colptr);
  u.h8[1] = *(const v8h*)(colptr + 8);
  return u.v;
}

// ---------------------------------------------------------------------------
// Kernel 0: f32 -> f16 weight conversion
__global__ void k_convert(const float* __restrict__ Wq, const float* __restrict__ Wp,
                          _Float16* __restrict__ WqH, _Float16* __restrict__ WpH) {
  int i = blockIdx.x * blockDim.x + threadIdx.x;
  if (i < QKVO * CCH) WqH[i] = (_Float16)Wq[i];
  else if (i < QKVO * CCH + CCH * CCH) {
    int j = i - QKVO * CCH;
    WpH[j] = (_Float16)Wp[j];
  }
}

// ---------------------------------------------------------------------------
// Kernel 1: qkvo = W_qkvo(1536x384) @ feat(384xN) + b.
// Block tile 64(M) x 32(N tokens); 4 waves, each wave = one 16-row A tile
// reused across TWO 16-token B tiles (B staged once in LDS for all 4 waves).
// grid = (320 token-tiles-of-32, 24 M-blocks), block = 128.
__global__ __launch_bounds__(128)
void k_qkvo(const float* __restrict__ z, const float* __restrict__ x,
            const _Float16* __restrict__ Wh, const float* __restrict__ bias,
            _Float16* __restrict__ qb, _Float16* __restrict__ kb,
            _Float16* __restrict__ vT, float* __restrict__ ob) {
  __shared__ _Float16 Bt[32 * 32];   // [token][k] f16

  int T = blockIdx.x;
  int b = T / 40, r = T % 40;
  const float* feat; int ldn, n0s, tokbase;
  if (r < 8) { feat = z + (size_t)b * CCH * NZ; ldn = NZ; n0s = r * 32;       tokbase = n0s; }
  else       { feat = x + (size_t)b * CCH * NX; ldn = NX; n0s = (r - 8) * 32; tokbase = NZ + n0s; }

  int tid = threadIdx.x, lane = tid & 31, wave = tid >> 5;
  int m0 = blockIdx.y * 64 + wave * 16;
  int col = lane & 15, kb2 = (lane >> 4) * 16;

  v8f acc0 = {}, acc1 = {};
  for (int k0 = 0; k0 < CCH; k0 += 32) {
    __syncthreads();
    // stage 32 channels x 32 tokens, f32->f16, transposed to [token][k]
    {
      int e0 = tid * 8;                    // 1024 elems / 128 threads
      int kk = e0 >> 5, tt = e0 & 31;      // 8 consecutive e stay in one row
      const float* src = feat + (size_t)(k0 + kk) * ldn + n0s + tt;
#pragma unroll
      for (int i = 0; i < 8; ++i) Bt[(tt + i) * 32 + kk] = (_Float16)src[i];
    }
    __syncthreads();

    v16h a  = load_a16x32(Wh + (size_t)m0 * CCH + k0, CCH, lane);
    v16h b0 = load_bcol(Bt + (col)      * 32 + kb2);
    v16h b1 = load_bcol(Bt + (16 + col) * 32 + kb2);
    acc0 = wmma_f16(a, b0, acc0);
    acc1 = wmma_f16(a, b1, acc1);
  }

  // Epilogue: scatter both token tiles to q / k / vT / o with bias
  int mb = (lane >> 4) * 8;
#pragma unroll
  for (int nt = 0; nt < 2; ++nt) {
    int tok = tokbase + nt * 16 + col;
    v8f acc = nt ? acc1 : acc0;
#pragma unroll
    for (int rr = 0; rr < 8; ++rr) {
      int o = m0 + mb + rr;
      float val = acc[rr] + bias[o];
      if (o < CCH) {                       // q, pre-scaled
        int c = o;
        qb[(((size_t)b * NHEAD + (c >> 5)) * NTOK + tok) * DHEAD + (c & 31)] =
            (_Float16)(val * SCALE);
      } else if (o < 2 * CCH) {            // k
        int c = o - CCH;
        kb[(((size_t)b * NHEAD + (c >> 5)) * NTOK + tok) * DHEAD + (c & 31)] = (_Float16)val;
      } else if (o < 3 * CCH) {            // v transposed [b][c][tok]
        int c = o - 2 * CCH;
        vT[((size_t)b * CCH + c) * NTOK + tok] = (_Float16)val;
      } else {                             // o gate f32
        int c = o - 3 * CCH;
        ob[((size_t)b * CCH + c) * NTOK + tok] = val;
      }
    }
  }
}

// ---------------------------------------------------------------------------
// Kernel 2: flash attention; one wave per (b, h, 16-query tile), 40 chunks of
// 32 keys. Softmax split across half-waves (each half owns 16 of a row's 32
// columns), combined through LDS with s_wait_dscnt fences.
__global__ __launch_bounds__(32)
void k_attn(const _Float16* __restrict__ qb, const _Float16* __restrict__ kb,
            const _Float16* __restrict__ vT, float* __restrict__ aout) {
  __shared__ float    Sm[16 * 32];
  __shared__ float    rowsc[16];
  __shared__ float    halfred[2 * 16];   // [half][row]
  __shared__ _Float16 Pm[16 * 32];

  int bid = blockIdx.x;
  int qt = bid % 80;
  int h  = (bid / 80) % NHEAD;
  int b  = bid / (80 * NHEAD);
  int n0 = qt * 16;
  int lane = threadIdx.x & 31;
  int col = lane & 15, hf = lane >> 4, mb = hf * 8, kb2 = hf * 16;

  const _Float16* qp    = qb + (((size_t)b * NHEAD + h) * NTOK + n0) * DHEAD;
  const _Float16* kbase = kb + (((size_t)b * NHEAD + h) * NTOK) * DHEAD;
  const _Float16* vbase = vT + ((size_t)b * CCH + h * DHEAD) * NTOK;

  v16h qa = load_a16x32(qp, DHEAD, lane);
  v8f o0 = {}, o1 = {};
  float m_i = -1e30f, l_i = 0.f;   // replicated identically in both lanes of a row

  for (int mc = 0; mc < NTOK; mc += 32) {
    // ---- S = Q K^T ----
    v8f zc = {};
    v16h k0t = load_bcol(kbase + (size_t)(mc + col) * DHEAD + kb2);
    v16h k1t = load_bcol(kbase + (size_t)(mc + 16 + col) * DHEAD + kb2);
    v8f s0 = wmma_f16(qa, k0t, zc);
    v8f s1 = wmma_f16(qa, k1t, zc);

    // ---- stage S (D layout: lane=col, vgpr r -> row mb+r) ----
#pragma unroll
    for (int rr = 0; rr < 8; ++rr) {
      Sm[(mb + rr) * 32 + col]      = s0[rr];
      Sm[(mb + rr) * 32 + 16 + col] = s1[rr];
    }
    lds_fence();

    // ---- online softmax: lane owns (row = col, columns hf*16..hf*16+15) ----
    const float* srow = &Sm[col * 32 + hf * 16];
    float hmax = srow[0];
#pragma unroll
    for (int j = 1; j < 16; ++j) hmax = fmaxf(hmax, srow[j]);
    halfred[hf * 16 + col] = hmax;
    lds_fence();
    float omax = halfred[(1 - hf) * 16 + col];
    float mnew = fmaxf(m_i, fmaxf(hmax, omax));
    float sc   = __expf(m_i - mnew);
    float hsum = 0.f;
    _Float16* prow = &Pm[col * 32 + hf * 16];
#pragma unroll
    for (int j = 0; j < 16; ++j) {
      float p = __expf(srow[j] - mnew);
      hsum += p;
      prow[j] = (_Float16)p;
    }
    halfred[hf * 16 + col] = hsum;   // ordered after omax load by data dep
    lds_fence();
    float osum = halfred[(1 - hf) * 16 + col];
    l_i = l_i * sc + hsum + osum;
    m_i = mnew;
    if (hf == 0) rowsc[col] = sc;
    lds_fence();

    // ---- rescale accumulators ----
#pragma unroll
    for (int rr = 0; rr < 8; ++rr) {
      float s = rowsc[mb + rr];
      o0[rr] *= s; o1[rr] *= s;
    }

    // ---- O += P V ----
    v16h pa = load_a16x32((const _Float16*)Pm, 32, lane);
    v16h v0t = load_bcol(vbase + (size_t)(col)      * NTOK + mc + kb2);
    v16h v1t = load_bcol(vbase + (size_t)(16 + col) * NTOK + mc + kb2);
    o0 = wmma_f16(pa, v0t, o0);
    o1 = wmma_f16(pa, v1t, o1);
    lds_fence();   // drain Pm/Sm reads before next chunk overwrites
  }

  // ---- normalize and store f32 ----
  if (hf == 0) rowsc[col] = 1.0f / l_i;
  lds_fence();
  float* op = aout + (((size_t)b * NHEAD + h) * NTOK + n0) * DHEAD;
#pragma unroll
  for (int rr = 0; rr < 8; ++rr) {
    float inv = rowsc[mb + rr];
    op[(mb + rr) * DHEAD + col]      = o0[rr] * inv;
    op[(mb + rr) * DHEAD + 16 + col] = o1[rr] * inv;
  }
}

// ---------------------------------------------------------------------------
// Kernel 3: lepe (depthwise 5x5 on v) + (attn + lepe) * o, token-major f16 out
__global__ void k_lepe_gate(const _Float16* __restrict__ vT, const float* __restrict__ aout,
                            const float* __restrict__ ob, const float* __restrict__ Wl,
                            const float* __restrict__ bl, _Float16* __restrict__ u) {
  int idx = blockIdx.x * blockDim.x + threadIdx.x;
  if (idx >= BATCH * CCH * NTOK) return;
  int tok = idx % NTOK;
  int bc  = idx / NTOK;
  int c   = bc % CCH;
  int b   = bc / CCH;

  int Hs, hh, ww, tok0;
  if (tok < NZ) { Hs = 16; tok0 = 0;  hh = tok >> 4; ww = tok & 15; }
  else          { Hs = 32; tok0 = NZ; int t2 = tok - NZ; hh = t2 >> 5; ww = t2 & 31; }

  const _Float16* vrow = vT + ((size_t)b * CCH + c) * NTOK + tok0;
  const float* wl = Wl + c * 25;
  float acc = bl[c];
  for (int i = 0; i < 5; ++i) {
    int y = hh + i - 2;
    if ((unsigned)y >= (unsigned)Hs) continue;
    for (int j = 0; j < 5; ++j) {
      int xx = ww + j - 2;
      if ((unsigned)xx >= (unsigned)Hs) continue;
      acc += wl[i * 5 + j] * (float)vrow[y * Hs + xx];
    }
  }
  int h2 = c >> 5, dd = c & 31;
  float att = aout[(((size_t)b * NHEAD + h2) * NTOK + tok) * DHEAD + dd];
  float val = (att + acc) * ob[idx];
  u[((size_t)b * NTOK + tok) * CCH + c] = (_Float16)val;
}

// ---------------------------------------------------------------------------
// Kernel 4: out = W_proj @ u + b. 2x2 register-blocked: one wave computes a
// 32(out-ch) x 32(token) tile, 4 WMMAs per K-step, no LDS (u is token-major).
// grid = (320 token-tiles-of-32, 12 M-tiles-of-32), block = 32.
__global__ __launch_bounds__(32)
void k_proj(const _Float16* __restrict__ Wp, const _Float16* __restrict__ u,
            const float* __restrict__ bias, float* __restrict__ out) {
  int T = blockIdx.x;
  int b = T / 40, r = T % 40;
  int tokbase = r * 32;
  int m0 = blockIdx.y * 32;
  int lane = threadIdx.x & 31;
  int col = lane & 15, mb = (lane >> 4) * 8, kb2 = (lane >> 4) * 16;

  const _Float16* uc0 = u + ((size_t)b * NTOK + tokbase + col) * CCH + kb2;
  const _Float16* uc1 = u + ((size_t)b * NTOK + tokbase + 16 + col) * CCH + kb2;
  v8f a00 = {}, a01 = {}, a10 = {}, a11 = {};
  for (int k0 = 0; k0 < CCH; k0 += 32) {
    v16h wa0 = load_a16x32(Wp + (size_t)m0 * CCH + k0, CCH, lane);
    v16h wa1 = load_a16x32(Wp + (size_t)(m0 + 16) * CCH + k0, CCH, lane);
    v16h ub0 = load_bcol(uc0 + k0);
    v16h ub1 = load_bcol(uc1 + k0);
    a00 = wmma_f16(wa0, ub0, a00);
    a01 = wmma_f16(wa0, ub1, a01);
    a10 = wmma_f16(wa1, ub0, a10);
    a11 = wmma_f16(wa1, ub1, a11);
  }

#pragma unroll
  for (int mi = 0; mi < 2; ++mi) {
#pragma unroll
    for (int ni = 0; ni < 2; ++ni) {
      v8f acc = mi ? (ni ? a11 : a10) : (ni ? a01 : a00);
      int tok = tokbase + ni * 16 + col;
#pragma unroll
      for (int rr = 0; rr < 8; ++rr) {
        int oc = m0 + mi * 16 + mb + rr;
        float val = acc[rr] + bias[oc];
        if (tok < NZ)
          out[((size_t)b * CCH + oc) * NZ + tok] = val;
        else
          out[(size_t)BATCH * CCH * NZ + ((size_t)b * CCH + oc) * NX + (tok - NZ)] = val;
      }
    }
  }
}

// ---------------------------------------------------------------------------
extern "C" void kernel_launch(void* const* d_in, const int* in_sizes, int n_in,
                              void* d_out, int out_size, void* d_ws, size_t ws_size,
                              hipStream_t stream) {
  const float* z      = (const float*)d_in[0];
  const float* x      = (const float*)d_in[1];
  const float* W_qkvo = (const float*)d_in[2];
  const float* b_qkvo = (const float*)d_in[3];
  const float* W_lepe = (const float*)d_in[4];
  const float* b_lepe = (const float*)d_in[5];
  const float* W_proj = (const float*)d_in[6];
  const float* b_proj = (const float*)d_in[7];
  float* out = (float*)d_out;

  char* ws = (char*)d_ws;
  _Float16* WqH  = (_Float16*)(ws + OFF_WQH);
  _Float16* WpH  = (_Float16*)(ws + OFF_WPH);
  _Float16* qbuf = (_Float16*)(ws + OFF_Q);
  _Float16* kbuf = (_Float16*)(ws + OFF_K);
  _Float16* vT   = (_Float16*)(ws + OFF_VT);
  float*    ob   = (float*)   (ws + OFF_O);
  float*    aout = (float*)   (ws + OFF_AOUT);
  _Float16* ubuf = (_Float16*)(ws + OFF_U);

  k_convert<<<(QKVO * CCH + CCH * CCH + 255) / 256, 256, 0, stream>>>(W_qkvo, W_proj, WqH, WpH);

  k_qkvo<<<dim3(320, 24), 128, 0, stream>>>(z, x, WqH, b_qkvo, qbuf, kbuf, vT, ob);

  k_attn<<<BATCH * NHEAD * 80, 32, 0, stream>>>(qbuf, kbuf, vT, aout);

  k_lepe_gate<<<(BATCH * CCH * NTOK + 255) / 256, 256, 0, stream>>>(vT, aout, ob,
                                                                    W_lepe, b_lepe, ubuf);

  k_proj<<<dim3(320, 12), 32, 0, stream>>>(WpH, ubuf, b_proj, out);
}
